// RNNDec_38019050505011
// MI455X (gfx1250) — matrix operand order repaired and checked by reading, not compiled, not run on hardware
//
#include <hip/hip_runtime.h>
#include <hip/hip_bf16.h>

// ---------------------------------------------------------------------------
// GRU decoder for MI455X (gfx1250, wave32, WMMA).
//   B=16, T=2048, H=1024.
// Phase 1: gi = h_enc @ w_ih^T + b_ih, v_wmma_f32_16x16x32_bf16, 4x4 register
//          blocking per wave (16 accumulators) to cut L2 traffic 4x.
// Phase 2: persistent 32-WG scan; each 16-col j-tile is computed by a PAIR of
//          waves splitting the K=1024 reduction (48 WMMA each); partial sums
//          are combined through LDS; even wave runs the gate epilogue.
//          Double-buffered h (f32 + bf16), device-scope atomic grid barrier.
// ---------------------------------------------------------------------------

typedef __bf16 bf16_t;
typedef bf16_t v16bf __attribute__((ext_vector_type(16)));
typedef bf16_t v8bf  __attribute__((ext_vector_type(8)));
typedef float  v8f   __attribute__((ext_vector_type(8)));

#define B_DIM   16
#define T_DIM   2048
#define H_DIM   1024
#define G3_DIM  3072
#define BT_DIM  32768   // B*T

// Workspace layout (bytes, all 256-aligned):
#define OFF_WIH 0ull          // w_ih bf16: 6291456
#define OFF_WHH 6291456ull    // w_hh bf16: 6291456
#define OFF_GI  12582912ull   // gi f32:    402653184
#define OFF_HBF 415236096ull  // h bf16 x2: 65536
#define OFF_HF  415301632ull  // h f32  x2: 131072
#define OFF_CNT 415432704ull  // barrier counter

__device__ __forceinline__ v8f wmma_bf16(v16bf a, v16bf b, v8f c) {
  // D = A(16x32 bf16) * B(32x16 bf16) + C(16x16 f32)
  return __builtin_amdgcn_wmma_f32_16x16x32_bf16(false, a, false, b, (short)0, c,
                                                 false, false);
}

// A-fragment (16x32 bf16, row-major source, leading dim lda):
// lane L: row m = L&15; K-runs {s..s+7} and {s+16..s+23}, s = (L&16)?8:0.
__device__ __forceinline__ v16bf load_a_bf16(const bf16_t* __restrict__ base,
                                             int lda, int lane) {
  const int m = lane & 15;
  const int s = (lane & 16) ? 8 : 0;
  const bf16_t* p = base + (size_t)m * lda + s;
  v8bf lo = *(const v8bf*)(p);
  v8bf hi = *(const v8bf*)(p + 16);
  return __builtin_shufflevector(lo, hi, 0, 1, 2, 3, 4, 5, 6, 7,
                                 8, 9, 10, 11, 12, 13, 14, 15);
}

// Same A-fragment from an f32 source, converting to bf16 in-register.
__device__ __forceinline__ v16bf load_a_f32cvt(const float* __restrict__ base,
                                               int lda, int lane) {
  const int m = lane & 15;
  const int s = (lane & 16) ? 8 : 0;
  const float* p = base + (size_t)m * lda + s;
  float4 a0 = *(const float4*)(p);
  float4 a1 = *(const float4*)(p + 4);
  float4 b0 = *(const float4*)(p + 16);
  float4 b1 = *(const float4*)(p + 20);
  v16bf v;
  v[0]  = (bf16_t)a0.x; v[1]  = (bf16_t)a0.y; v[2]  = (bf16_t)a0.z; v[3]  = (bf16_t)a0.w;
  v[4]  = (bf16_t)a1.x; v[5]  = (bf16_t)a1.y; v[6]  = (bf16_t)a1.z; v[7]  = (bf16_t)a1.w;
  v[8]  = (bf16_t)b0.x; v[9]  = (bf16_t)b0.y; v[10] = (bf16_t)b0.z; v[11] = (bf16_t)b0.w;
  v[12] = (bf16_t)b1.x; v[13] = (bf16_t)b1.y; v[14] = (bf16_t)b1.z; v[15] = (bf16_t)b1.w;
  return v;
}

// B-fragment (32x16 bf16). B[k][n] = W[nbase+n][kb+k], W row-major [*, ldw]:
// lane L: col n = L&15; K-runs {s..s+7} and {s+16..s+23}, s = (L&16)?8:0.
__device__ __forceinline__ v16bf load_b_bf16(const bf16_t* __restrict__ w, int ldw,
                                             int nbase, int kb, int lane) {
  const int n = lane & 15;
  const int k = kb + ((lane & 16) ? 8 : 0);
  const bf16_t* p = w + (size_t)(nbase + n) * ldw + k;
  v8bf lo = *(const v8bf*)(p);
  v8bf hi = *(const v8bf*)(p + 16);
  return __builtin_shufflevector(lo, hi, 0, 1, 2, 3, 4, 5, 6, 7,
                                 8, 9, 10, 11, 12, 13, 14, 15);
}

__device__ __forceinline__ float fast_sigmoid(float x) {
  return __builtin_amdgcn_rcpf(1.0f + __expf(-x));
}
__device__ __forceinline__ float fast_tanh(float x) {
  // tanh(x) = 1 - 2/(e^{2x}+1); saturates correctly at +-1.
  return 1.0f - 2.0f * __builtin_amdgcn_rcpf(1.0f + __expf(2.0f * x));
}

// --------------------------- helper kernels --------------------------------

__global__ void __launch_bounds__(256)
gru_cvt_bf16(const float* __restrict__ in, bf16_t* __restrict__ out, int n) {
  const int stride = gridDim.x * blockDim.x;
  for (int i = blockIdx.x * blockDim.x + threadIdx.x; i < n; i += stride)
    out[i] = (bf16_t)in[i];
}

__global__ void __launch_bounds__(256)
gru_scan_init(bf16_t* __restrict__ hbf, float* __restrict__ hf,
              unsigned* __restrict__ cnt) {
  const int i = blockIdx.x * blockDim.x + threadIdx.x;
  const int n = 2 * B_DIM * H_DIM;
  const int stride = gridDim.x * blockDim.x;
  for (int k = i; k < n; k += stride) {
    hbf[k] = (bf16_t)0.0f;
    hf[k] = 0.0f;
  }
  if (i == 0) *cnt = 0u;
}

// ------------------------ phase 1: gi GEMM ---------------------------------
// 4 M-tiles x 4 N-tiles per wave (16 f32 accumulators = 128 VGPRs).
#define G1_TPB  256
#define G1_BLKS 768
#define MT_BLK  4
#define NT_BLK  4

__global__ void __launch_bounds__(G1_TPB)
gru_gemm_ih(const float* __restrict__ x,      // [32768][1024] f32 (h_enc)
            const bf16_t* __restrict__ wb,    // [3072][1024] bf16 (w_ih)
            const float* __restrict__ b_ih,   // [3072]
            float* __restrict__ gi) {         // [32768][3072] f32
  const int lane = threadIdx.x & 31;
  const int wave = blockIdx.x * (G1_TPB / 32) + (threadIdx.x >> 5);
  const int nwaves = gridDim.x * (G1_TPB / 32);
  const int ngroups = G3_DIM / (16 * NT_BLK);        // 48
  const int mgroups = BT_DIM / (16 * MT_BLK);        // 512
  const int ntasks = mgroups * ngroups;              // 24576

  for (int task = wave; task < ntasks; task += nwaves) {
    const int mg = task / ngroups;
    const int ng = task % ngroups;
    const int r0 = mg * (16 * MT_BLK);               // first output row
    const int n0 = ng * (16 * NT_BLK);               // first output col

    v8f acc[MT_BLK][NT_BLK];
#pragma unroll
    for (int i = 0; i < MT_BLK; ++i)
#pragma unroll
      for (int j = 0; j < NT_BLK; ++j) acc[i][j] = (v8f){0, 0, 0, 0, 0, 0, 0, 0};

    for (int kb = 0; kb < H_DIM; kb += 32) {
      v16bf a[MT_BLK];
#pragma unroll
      for (int mt = 0; mt < MT_BLK; ++mt)
        a[mt] = load_a_f32cvt(x + (size_t)(r0 + mt * 16) * H_DIM + kb, H_DIM, lane);
#pragma unroll
      for (int nt = 0; nt < NT_BLK; ++nt) {
        v16bf b = load_b_bf16(wb, H_DIM, n0 + nt * 16, kb, lane);
#pragma unroll
        for (int mt = 0; mt < MT_BLK; ++mt)
          acc[mt][nt] = wmma_bf16(a[mt], b, acc[mt][nt]);
      }
    }

    // C/D layout: VGPR e holds M = e (lanes 0-15) / e+8 (lanes 16-31), N = lane&15.
    const int m0 = (lane & 16) ? 8 : 0;
    const int nn = lane & 15;
#pragma unroll
    for (int nt = 0; nt < NT_BLK; ++nt) {
      const int col = n0 + nt * 16 + nn;
      const float bi = b_ih[col];
#pragma unroll
      for (int mt = 0; mt < MT_BLK; ++mt) {
#pragma unroll
        for (int e = 0; e < 8; ++e) {
          const int row = r0 + mt * 16 + m0 + e;
          gi[(size_t)row * G3_DIM + col] = acc[mt][nt][e] + bi;
        }
      }
    }
  }
}

// ------------------------ phase 2: recurrent scan --------------------------
// 32 WGs x 128 threads = 128 waves. Wave pair (2p, 2p+1) owns j-tile
// blockIdx*2+p; even wave reduces K [0,512), odd wave K [512,1024).
// Odd wave's partial accumulators cross through LDS; even wave does gates.
#define SCAN_WGS 32
#define SCAN_TPB 128
#define KSPLIT   512

__global__ void __launch_bounds__(SCAN_TPB)
gru_scan(const float* __restrict__ gi,        // [32768][3072] f32
         const bf16_t* __restrict__ whh,      // [3072][1024] bf16
         const float* __restrict__ b_hh,      // [3072]
         float* __restrict__ out,             // [16][2048][1024] f32
         bf16_t* __restrict__ hbf,            // [2][16][1024] bf16
         float* __restrict__ hf,              // [2][16][1024] f32
         unsigned* __restrict__ cnt) {
  __shared__ v8f lds_acc[2][3][32];           // [pair][gate][lane]

  const int lane = threadIdx.x & 31;
  const int wv = threadIdx.x >> 5;            // 0..3
  const int pair = wv >> 1;                   // 0..1
  const int kh = wv & 1;                      // 0 = low K + epilogue, 1 = high K
  const int jt = blockIdx.x * 2 + pair;       // 0..63
  const int j0 = jt * 16;
  const int m0 = (lane & 16) ? 8 : 0;
  const int nn = lane & 15;
  const int col = j0 + nn;
  const int kbeg = kh * KSPLIT;

  const float bhr = b_hh[col];
  const float bhz = b_hh[H_DIM + col];
  const float bhn = b_hh[2 * H_DIM + col];

#pragma unroll 1
  for (int t = 0; t < T_DIM; ++t) {
    const int rd = t & 1;
    const int wr = rd ^ 1;
    const bf16_t* hA = hbf + rd * (B_DIM * H_DIM);

    v8f ar = (v8f){0, 0, 0, 0, 0, 0, 0, 0};
    v8f az = (v8f){0, 0, 0, 0, 0, 0, 0, 0};
    v8f an_ = (v8f){0, 0, 0, 0, 0, 0, 0, 0};

    for (int kb = kbeg; kb < kbeg + KSPLIT; kb += 32) {
      v16bf a = load_a_bf16(hA + kb, H_DIM, lane);
      v16bf br = load_b_bf16(whh, H_DIM, j0, kb, lane);
      v16bf bz = load_b_bf16(whh, H_DIM, H_DIM + j0, kb, lane);
      v16bf bn = load_b_bf16(whh, H_DIM, 2 * H_DIM + j0, kb, lane);
      ar = wmma_bf16(a, br, ar);
      az = wmma_bf16(a, bz, az);
      an_ = wmma_bf16(a, bn, an_);
    }

    // Odd wave ships partial sums to its pair's even wave via LDS.
    if (kh == 1) {
      lds_acc[pair][0][lane] = ar;
      lds_acc[pair][1][lane] = az;
      lds_acc[pair][2][lane] = an_;
    }
    __syncthreads();

    if (kh == 0) {
      ar += lds_acc[pair][0][lane];
      az += lds_acc[pair][1][lane];
      an_ += lds_acc[pair][2][lane];

      const float* hprev = hf + rd * (B_DIM * H_DIM);
      float* hnext = hf + wr * (B_DIM * H_DIM);
      bf16_t* hnb = hbf + wr * (B_DIM * H_DIM);

#pragma unroll
      for (int e = 0; e < 8; ++e) {
        const int bb = m0 + e;
        const size_t gbase = ((size_t)bb * T_DIM + t) * G3_DIM;
        const float ir = __builtin_nontemporal_load(gi + gbase + col);
        const float iz = __builtin_nontemporal_load(gi + gbase + H_DIM + col);
        const float in_ = __builtin_nontemporal_load(gi + gbase + 2 * H_DIM + col);
        const float hr = ar[e] + bhr;
        const float hz = az[e] + bhz;
        const float hn = an_[e] + bhn;
        const float rg = fast_sigmoid(ir + hr);
        const float zg = fast_sigmoid(iz + hz);
        const float ng = fast_tanh(in_ + rg * hn);
        const float hp = hprev[(size_t)bb * H_DIM + col];
        const float hv = (1.0f - zg) * ng + zg * hp;
        __builtin_nontemporal_store(hv, out + ((size_t)bb * T_DIM + t) * H_DIM + col);
        hnext[(size_t)bb * H_DIM + col] = hv;
        hnb[(size_t)bb * H_DIM + col] = (bf16_t)hv;
      }
    } else if (t + 1 < T_DIM) {
      // Pull next timestep's gi tile toward L2 while the pair finishes.
      const size_t pf = ((size_t)m0 * T_DIM + (t + 1)) * G3_DIM + col;
      __builtin_prefetch(gi + pf, 0, 1);
    }

    // Device-scope grid barrier (monotonic counter).
    __syncthreads();
    if (threadIdx.x == 0) {
      __builtin_amdgcn_fence(__ATOMIC_RELEASE, "agent");
      __hip_atomic_fetch_add(cnt, 1u, __ATOMIC_RELAXED, __HIP_MEMORY_SCOPE_AGENT);
      const unsigned target = (unsigned)(t + 1) * SCAN_WGS;
      while (__hip_atomic_load(cnt, __ATOMIC_RELAXED, __HIP_MEMORY_SCOPE_AGENT) <
             target)
        __builtin_amdgcn_s_sleep(1);
      __builtin_amdgcn_fence(__ATOMIC_ACQUIRE, "agent");
    }
    __syncthreads();
  }
}

// ------------------------------ launch -------------------------------------

extern "C" void kernel_launch(void* const* d_in, const int* in_sizes, int n_in,
                              void* d_out, int out_size, void* d_ws, size_t ws_size,
                              hipStream_t stream) {
  const float* h_enc = (const float*)d_in[0];   // [16,2048,1024]
  const float* w_ih = (const float*)d_in[1];    // [3072,1024]
  const float* w_hh = (const float*)d_in[2];    // [3072,1024]
  const float* b_ih = (const float*)d_in[3];    // [3072]
  const float* b_hh = (const float*)d_in[4];    // [3072]
  float* out = (float*)d_out;                   // [16,2048,1024]

  char* ws = (char*)d_ws;
  bf16_t* wih_b = (bf16_t*)(ws + OFF_WIH);
  bf16_t* whh_b = (bf16_t*)(ws + OFF_WHH);
  float* gi = (float*)(ws + OFF_GI);
  bf16_t* hbf = (bf16_t*)(ws + OFF_HBF);
  float* hf = (float*)(ws + OFF_HF);
  unsigned* cnt = (unsigned*)(ws + OFF_CNT);

  const int wsz = G3_DIM * H_DIM;  // 3145728 weight elements

  gru_cvt_bf16<<<1024, 256, 0, stream>>>(w_ih, wih_b, wsz);
  gru_cvt_bf16<<<1024, 256, 0, stream>>>(w_hh, whh_b, wsz);
  gru_scan_init<<<128, 256, 0, stream>>>(hbf, hf, cnt);
  gru_gemm_ih<<<G1_BLKS, G1_TPB, 0, stream>>>(h_enc, wih_b, b_ih, gi);
  gru_scan<<<SCAN_WGS, SCAN_TPB, 0, stream>>>(gi, whh_b, b_hh, out, hbf, hf, cnt);
}
